// Decoder_52759378264144
// MI455X (gfx1250) — matrix operand order repaired and checked by reading, not compiled, not run on hardware
//
#include <hip/hip_runtime.h>
#include <hip/hip_bf16.h>

typedef __bf16 bf16_t;
typedef __attribute__((ext_vector_type(16))) __bf16 v16bf;
typedef __attribute__((ext_vector_type(8)))  float  v8f;

#define BB 16
#define T_ENC 256
#define T_MEL 256
#define EH 512
#define ATTN 128
#define PRE_OUT 256
#define MEL 80
#define LSTM_DIM 1024
#define PF 512
#define CH 80
#define CW 256
#define HW (CH*CW)
#define HP 84           // CH + 4 (pad 2 each side)
#define WPAD 272        // >= CW + 4, multiple of 16 for aligned rows

union Frag16 { v16bf v; uint4 q[2]; };

__device__ __forceinline__ float sigm(float x) { return 1.0f / (1.0f + __expf(-x)); }

// ---------------------------------------------------------------------------
// WMMA GEMM: C[M,N] = act(A[M,K] * W[N,K]^T + bias).  REQUIRES K % 32 == 0.
// A fp32 staged once per block into LDS as bf16 (shared by all 8 waves);
// W bf16 [N,K] streamed from global, software-pipelined one iteration ahead.
// Launch with dynamic LDS = 16*K*2 bytes.
// ---------------------------------------------------------------------------
__global__ void k_gemm_af32(const float* __restrict__ A, const bf16_t* __restrict__ W,
                            const float* __restrict__ bias, float* __restrict__ C,
                            int M, int N, int K, int act)
{
    extern __shared__ bf16_t sA[];                  // [16][K]
    int tid  = threadIdx.x;
    int lane = tid & 31;
    int wv   = tid >> 5;
    int n0   = blockIdx.x * 128 + wv * 16;
    int m0   = blockIdx.y * 16;
    int l15  = lane & 15;
    int half = lane >> 4;

    // cooperative stage: A tile (16 x K) fp32 -> bf16 in LDS (float4 granular)
    int tot4 = (16 * K) >> 2;
    for (int i = tid; i < tot4; i += 256) {
        int idx = i << 2;                           // K%4==0 -> pack stays in-row
        int m = idx / K, k = idx % K;
        float4 v = *(const float4*)(A + (size_t)(m0 + m) * K + k);
        bf16_t* d = sA + m * K + k;
        d[0] = (bf16_t)v.x; d[1] = (bf16_t)v.y; d[2] = (bf16_t)v.z; d[3] = (bf16_t)v.w;
    }
    __syncthreads();

    // 16-bit A 16x32 layout (contiguous, matching B): lanes 0-15 hold K=0..15,
    // lanes 16-31 hold K=16..31 for their M row.
    const bf16_t* sa   = sA + l15 * K + (half << 4);
    const bf16_t* brow = W + (size_t)(n0 + l15) * K + (half << 4);

    Frag16 bu, bn, au;
    bu.q[0] = *(const uint4*)(brow);
    bu.q[1] = *(const uint4*)(brow + 8);

    v8f acc = {};
    for (int kt = 0; kt < K; kt += 32) {
        // issue next B fragment first (dead on last iteration, <=96B overrun in ws)
        bn.q[0] = *(const uint4*)(brow + kt + 32);
        bn.q[1] = *(const uint4*)(brow + kt + 40);
        au.q[0] = *(const uint4*)(sa + kt);         // ds_load_b128
        au.q[1] = *(const uint4*)(sa + kt + 8);
        __builtin_prefetch((const void*)(brow + kt + 160), 0, 1);  // speculative
        acc = __builtin_amdgcn_wmma_f32_16x16x32_bf16(false, au.v, false, bu.v,
                                                      (short)0, acc, false, false);
        bu = bn;
    }
    float bv = bias ? bias[n0 + l15] : 0.0f;
#pragma unroll
    for (int r = 0; r < 8; ++r) {
        int m = m0 + r + half * 8;                  // C layout: VGPR r -> row
        float v = acc[r] + bv;
        if (act == 1) v = fmaxf(v, 0.0f);
        C[(size_t)m * N + n0 + l15] = v;
    }
}

// ---------------------------------------------------------------------------
// Implicit-GEMM 5x5 conv, 512->512, bf16. Input ZERO-PADDED [16,512,84,272].
// The 512x5x20 input patch is staged in LDS (100KB), reused by all 8 waves.
// Weights are PRE-PERMUTED to k' = tap*512 + ci, so each 32-wide K-chunk has
// a uniform tap: the 16-element A gather becomes immediate-offset ds loads
// off one base pointer (offsets are multiples of 100 elems = 200 bytes).
// Launch with dynamic LDS = 512*5*20*2 bytes.
// ---------------------------------------------------------------------------
__global__ void k_conv5x5_wmma(const bf16_t* __restrict__ XP, const bf16_t* __restrict__ W,
                               const float* __restrict__ bias, bf16_t* __restrict__ Y)
{
    extern __shared__ bf16_t sP[];                  // [512][5][20]
    const int KTOT = PF * 25;
    int tid  = threadIdx.x;
    int lane = tid & 31;
    int wv   = tid >> 5;
    int co0  = blockIdx.x * 128 + wv * 16;
    int h    = blockIdx.y >> 4;
    int w0   = (blockIdx.y & 15) << 4;
    int b    = blockIdx.z;
    int l15  = lane & 15;
    int half = lane >> 4;

    const bf16_t* xb = XP + (size_t)b * PF * HP * WPAD;
    {   // stage patch as u32 words (2 bf16 each): 25600 words
        uint* sw = (uint*)sP;
        const int WORDS = PF * 5 * 10;
        for (int wi = tid; wi < WORDS; wi += 256) {
            int ci = wi / 50, rem = wi % 50, dh = rem / 10, xw = rem % 10;
            const uint* gp = (const uint*)(xb + ((size_t)ci * HP + h + dh) * WPAD + w0);
            sw[wi] = gp[xw];
        }
    }
    __syncthreads();

    const bf16_t* brow = W + (size_t)(co0 + l15) * KTOT + (half << 4);
    Frag16 bu, bn, au;
    bu.q[0] = *(const uint4*)(brow);
    bu.q[1] = *(const uint4*)(brow + 8);

    v8f acc = {};
    for (int kt = 0; kt < KTOT; kt += 32) {
        bn.q[0] = *(const uint4*)(brow + kt + 32);
        bn.q[1] = *(const uint4*)(brow + kt + 40);
        // k' = tap*512 + ci: tap uniform across this chunk
        int r  = kt >> 9;
        int dh = r / 5, dw = r - dh * 5;
        const bf16_t* sb = sP + (kt & 511) * 100 + dh * 20 + dw + l15 + half * 1600;
#pragma unroll
        for (int e = 0; e < 16; ++e) {
            int off = e;                            // ci offset (half folded into sb)
            au.v[e] = sb[off * 100];                // ds_load_u16, immediate offset
        }
        __builtin_prefetch((const void*)(brow + kt + 160), 0, 1);
        acc = __builtin_amdgcn_wmma_f32_16x16x32_bf16(false, au.v, false, bu.v,
                                                      (short)0, acc, false, false);
        bu = bn;
    }
    float bv = bias[co0 + l15];
#pragma unroll
    for (int r = 0; r < 8; ++r) {
        int w = w0 + r + half * 8;
        Y[(((size_t)b * PF + (co0 + l15)) * CH + h) * CW + w] = (bf16_t)(acc[r] + bv);
    }
}

// ---------------------------------------------------------------------------
// Attention step (scalar, latency-bound). One block per batch row.
// ---------------------------------------------------------------------------
__global__ void k_attn_step(const float* __restrict__ z1,
                            const float* __restrict__ qpW, const float* __restrict__ qpB,
                            const float* __restrict__ qW,  const float* __restrict__ qB,
                            const float* __restrict__ kvbuf,
                            const float* __restrict__ locw,   // [31] + bias at [31]
                            const float* __restrict__ cumW, const float* __restrict__ cumB,
                            const float* __restrict__ pre_out,
                            float* __restrict__ pa, float* __restrict__ pe,
                            float* __restrict__ z0, int t)
{
    __shared__ float sh1[LSTM_DIM], q1[ATTN], q2[ATTN];
    __shared__ float spa[T_ENC], spe[T_ENC], se[T_ENC], sred[T_ENC];
    int b = blockIdx.x, tid = threadIdx.x;

    for (int i = tid; i < LSTM_DIM; i += 256) sh1[i] = z1[b * 2048 + 1024 + i];
    spa[tid] = pa[b * T_ENC + tid];
    spe[tid] = pe[b * T_ENC + tid];
    __syncthreads();

    if (tid < ATTN) {
        float s = qpB[tid];
        for (int d = 0; d < LSTM_DIM; ++d) s += qpW[tid * LSTM_DIM + d] * sh1[d];
        q1[tid] = s;
    }
    __syncthreads();
    if (tid < ATTN) {
        float s = qB[tid];
        for (int d = 0; d < ATTN; ++d) s += qW[tid * ATTN + d] * q1[d];
        q2[tid] = s;
    }
    __syncthreads();

    {
        const float* kr = kvbuf + (size_t)(b * T_ENC + tid) * 256;
        float s = 0.f;
        for (int a = 0; a < ATTN; ++a) s += q2[a] * kr[a];
        float loc = locw[31], cum = cumB[0];
        for (int k = 0; k < 31; ++k) {
            int tt = tid + k - 15;
            if (tt >= 0 && tt < T_ENC) { loc += locw[k] * spa[tt]; cum += cumW[k] * spe[tt]; }
        }
        se[tid] = s + loc + cum;
    }
    __syncthreads();

    sred[tid] = se[tid];  __syncthreads();
    for (int off = 128; off > 0; off >>= 1) {
        if (tid < off) sred[tid] = fmaxf(sred[tid], sred[tid + off]);
        __syncthreads();
    }
    float mx = sred[0];
    __syncthreads();
    float ex = __expf(se[tid] - mx);
    sred[tid] = ex;  __syncthreads();
    for (int off = 128; off > 0; off >>= 1) {
        if (tid < off) sred[tid] += sred[tid + off];
        __syncthreads();
    }
    float at = ex / sred[0];
    __syncthreads();
    spa[tid] = at;
    __syncthreads();

    if (tid < ATTN) {
        float s = 0.f;
        for (int tt = 0; tt < T_ENC; ++tt)
            s += spa[tt] * kvbuf[(size_t)(b * T_ENC + tt) * 256 + 128 + tid];
        z0[b * 1408 + tid] = s;
    }
    z0[b * 1408 + 128 + tid] = pre_out[(size_t)(b * T_MEL + t) * PRE_OUT + tid];
    pa[b * T_ENC + tid] = at;
    pe[b * T_ENC + tid] = se[tid];
}

__global__ void k_cell0(const float* __restrict__ gates, float* __restrict__ c0,
                        float* __restrict__ z1, float* __restrict__ z0)
{
    int idx = blockIdx.x * 256 + threadIdx.x;       // 16*1024
    int b = idx >> 10, j = idx & 1023;
    const float* g = gates + (size_t)b * 4096;
    float c = sigm(g[1024 + j]) * c0[idx] + sigm(g[j]) * tanhf(g[2048 + j]);
    float h = sigm(g[3072 + j]) * tanhf(c);
    c0[idx] = c;
    z1[b * 2048 + j] = h;
    z0[b * 1408 + 384 + j] = h;
}

__global__ void k_cell1_mel(const float* __restrict__ gates, float* __restrict__ c1,
                            float* __restrict__ z1,
                            const float* __restrict__ melW, const float* __restrict__ melB,
                            float* __restrict__ mel_out, int t)
{
    __shared__ float sh[LSTM_DIM];
    int b = blockIdx.x, tid = threadIdx.x;
    const float* g = gates + (size_t)b * 4096;
    for (int j = tid; j < LSTM_DIM; j += 256) {
        float c = sigm(g[1024 + j]) * c1[b * 1024 + j] + sigm(g[j]) * tanhf(g[2048 + j]);
        float h = sigm(g[3072 + j]) * tanhf(c);
        c1[b * 1024 + j] = c;
        z1[b * 2048 + 1024 + j] = h;
        sh[j] = h;
    }
    __syncthreads();
    if (tid < MEL) {
        float s = melB[tid];
        for (int d = 0; d < LSTM_DIM; ++d) s += melW[tid * LSTM_DIM + d] * sh[d];
        mel_out[(size_t)(b * T_MEL + t) * MEL + tid] = s;
    }
}

// ------------------------- postnet scalar kernels --------------------------
__global__ void k_conv_first(const float* __restrict__ mel, const bf16_t* __restrict__ W0,
                             const float* __restrict__ b0, bf16_t* __restrict__ Y)
{
    size_t total = (size_t)BB * PF * HW;
    for (size_t idx = (size_t)blockIdx.x * 256 + threadIdx.x; idx < total;
         idx += (size_t)gridDim.x * 256) {
        int w  = idx % CW;
        int h  = (idx / CW) % CH;
        int co = (idx / HW) % PF;
        int b  = idx / ((size_t)HW * PF);
        float s = b0[co];
        for (int kh = 0; kh < 5; ++kh) {
            int hh = h + kh - 2; if (hh < 0 || hh >= CH) continue;
            for (int kw = 0; kw < 5; ++kw) {
                int ww = w + kw - 2; if (ww < 0 || ww >= CW) continue;
                s += (float)W0[co * 25 + kh * 5 + kw] * mel[(size_t)(b * T_MEL + ww) * MEL + hh];
            }
        }
        Y[idx] = (bf16_t)s;
    }
}

// last conv reads the padded layer-3 activation: no bounds checks needed
__global__ void k_conv_last(const bf16_t* __restrict__ XP, const bf16_t* __restrict__ W4,
                            const float* __restrict__ b4, bf16_t* __restrict__ Y)
{
    int idx = blockIdx.x * 256 + threadIdx.x;       // 16*80*256
    int w = idx % CW, h = (idx / CW) % CH, b = idx / HW;
    const bf16_t* xb = XP + (size_t)b * PF * HP * WPAD;
    float s = b4[0];
    for (int ci = 0; ci < PF; ++ci)
        for (int kh = 0; kh < 5; ++kh) {
            const bf16_t* row = xb + ((size_t)ci * HP + h + kh) * WPAD + w;
            for (int kw = 0; kw < 5; ++kw)
                s += (float)W4[ci * 25 + kh * 5 + kw] * (float)row[kw];
        }
    Y[idx] = (bf16_t)s;
}

__global__ void k_bnstats(const bf16_t* __restrict__ X, float* __restrict__ stats, int C)
{
    __shared__ float ss[256], sq[256];
    int c = blockIdx.x, tid = threadIdx.x;
    float s = 0.f, q = 0.f;
    for (int i = tid; i < BB * HW; i += 256) {
        int b = i / HW, sp = i % HW;
        float v = (float)X[((size_t)(b * C + c)) * HW + sp];
        s += v; q += v * v;
    }
    ss[tid] = s; sq[tid] = q; __syncthreads();
    for (int off = 128; off > 0; off >>= 1) {
        if (tid < off) { ss[tid] += ss[tid + off]; sq[tid] += sq[tid + off]; }
        __syncthreads();
    }
    if (tid == 0) {
        float n = (float)(BB * HW);
        float mean = ss[0] / n;
        float var  = sq[0] / n - mean * mean;
        stats[c * 2] = mean;
        stats[c * 2 + 1] = rsqrtf(var + 1e-5f);
    }
}

// BN + tanh, writing into the zero-padded activation layout for the next conv
__global__ void k_bnact_pad(const bf16_t* __restrict__ X, bf16_t* __restrict__ YP,
                            const float* __restrict__ stats, const float* __restrict__ gam,
                            const float* __restrict__ bet)
{
    size_t total = (size_t)BB * PF * HW;
    for (size_t idx = (size_t)blockIdx.x * 256 + threadIdx.x; idx < total;
         idx += (size_t)gridDim.x * 256) {
        int w = idx % CW;
        int h = (idx / CW) % CH;
        int c = (idx / HW) % PF;
        int b = idx / ((size_t)HW * PF);
        float v = ((float)X[idx] - stats[c * 2]) * stats[c * 2 + 1] * gam[c] + bet[c];
        YP[(((size_t)b * PF + c) * HP + h + 2) * WPAD + w + 2] = (bf16_t)tanhf(v);
    }
}

__global__ void k_final_add(const bf16_t* __restrict__ P, const float* __restrict__ stats,
                            const float* __restrict__ g4, const float* __restrict__ be4,
                            const float* __restrict__ mel_out, float* __restrict__ out2)
{
    int idx = blockIdx.x * 256 + threadIdx.x;       // [B,T,MEL]
    int m = idx % MEL, t = (idx / MEL) % T_MEL, b = idx / (MEL * T_MEL);
    float x = (float)P[((size_t)b * CH + m) * CW + t];
    float v = (x - stats[0]) * stats[1] * g4[0] + be4[0];
    out2[idx] = mel_out[idx] + v;
}

// ------------------------------ prep kernels -------------------------------
__global__ void k_cvt(const float* __restrict__ s, bf16_t* __restrict__ d, size_t n)
{
    for (size_t i = (size_t)blockIdx.x * 256 + threadIdx.x; i < n; i += (size_t)gridDim.x * 256)
        d[i] = (bf16_t)s[i];
}

// convert with zero-pad of the K (inner) dimension: [N,Ks] f32 -> [N,Kd] bf16
__global__ void k_cvt_pad(const float* __restrict__ s, bf16_t* __restrict__ d,
                          int N, int Ks, int Kd)
{
    size_t total = (size_t)N * Kd;
    for (size_t i = (size_t)blockIdx.x * 256 + threadIdx.x; i < total;
         i += (size_t)gridDim.x * 256) {
        int n = i / Kd, j = i % Kd;
        d[i] = (j < Ks) ? (bf16_t)s[(size_t)n * Ks + j] : (bf16_t)0.0f;
    }
}

// zero-pad K dim of fp32 activations: [N,Ks] -> [N,Kd]
__global__ void k_pad_rows(const float* __restrict__ s, float* __restrict__ d,
                           int N, int Ks, int Kd)
{
    size_t total = (size_t)N * Kd;
    for (size_t i = (size_t)blockIdx.x * 256 + threadIdx.x; i < total;
         i += (size_t)gridDim.x * 256) {
        int n = i / Kd, j = i % Kd;
        d[i] = (j < Ks) ? s[(size_t)n * Ks + j] : 0.0f;
    }
}

__global__ void k_pack_lstm(const float* __restrict__ wih, const float* __restrict__ whh,
                            bf16_t* __restrict__ dst, int KI, int KH)
{
    size_t KT = (size_t)(KI + KH), total = 4096 * KT;
    for (size_t idx = (size_t)blockIdx.x * 256 + threadIdx.x; idx < total;
         idx += (size_t)gridDim.x * 256) {
        int n = idx / KT, j = idx % KT;
        float v = (j < KI) ? wih[(size_t)n * KI + j] : whh[(size_t)n * KH + (j - KI)];
        dst[idx] = (bf16_t)v;
    }
}

// conv weight repack: src [co][ci][25] f32 -> dst [co][k'] bf16, k' = tap*512+ci
__global__ void k_pack_convw(const float* __restrict__ src, bf16_t* __restrict__ dst)
{
    size_t total = (size_t)PF * PF * 25;
    for (size_t i = (size_t)blockIdx.x * 256 + threadIdx.x; i < total;
         i += (size_t)gridDim.x * 256) {
        int co  = i / (PF * 25);
        int rem = i % (PF * 25);
        int r   = rem >> 9;                 // tap 0..24
        int ci  = rem & (PF - 1);
        dst[i] = (bf16_t)src[((size_t)co * PF + ci) * 25 + r];
    }
}

__global__ void k_add2(const float* a, const float* b, float* o, int n)
{
    int i = blockIdx.x * 256 + threadIdx.x;
    if (i < n) o[i] = a[i] + b[i];
}

__global__ void k_locmean(const float* __restrict__ lw, const float* __restrict__ lb,
                          float* __restrict__ out)
{
    int tid = threadIdx.x;
    if (tid < 31) {
        float s = 0.f;
        for (int c = 0; c < ATTN; ++c) s += lw[c * 31 + tid];
        out[tid] = s / (float)ATTN;
    } else if (tid == 31) {
        float s = 0.f;
        for (int c = 0; c < ATTN; ++c) s += lb[c];
        out[31] = s / (float)ATTN;
    }
}

__global__ void k_fill(float* p, float v, int n)
{
    int i = blockIdx.x * 256 + threadIdx.x;
    if (i < n) p[i] = v;
}

__global__ void k_fill_bf16(bf16_t* p, size_t n)
{
    for (size_t i = (size_t)blockIdx.x * 256 + threadIdx.x; i < n; i += (size_t)gridDim.x * 256)
        p[i] = (bf16_t)0.0f;
}

// ---------------------------------------------------------------------------
extern "C" void kernel_launch(void* const* d_in, const int* in_sizes, int n_in,
                              void* d_out, int out_size, void* d_ws, size_t ws_size,
                              hipStream_t stream)
{
    // inputs in setup_inputs() insertion order, params flattened leaf-by-leaf
    const float* x       = (const float*)d_in[0];
    const float* mel_seq = (const float*)d_in[1];
    const float* preW1 = (const float*)d_in[2];
    const float* preB1 = (const float*)d_in[3];
    const float* preW2 = (const float*)d_in[4];
    const float* preB2 = (const float*)d_in[5];
    const float* locW  = (const float*)d_in[6];
    const float* locB  = (const float*)d_in[7];
    const float* cumW  = (const float*)d_in[8];
    const float* cumB  = (const float*)d_in[9];
    const float* qpW   = (const float*)d_in[10];
    const float* qpB   = (const float*)d_in[11];
    const float* kvpW  = (const float*)d_in[12];
    const float* kvpB  = (const float*)d_in[13];
    const float* qW    = (const float*)d_in[14];
    const float* qB    = (const float*)d_in[15];
    const float* kvW   = (const float*)d_in[16];
    const float* kvB   = (const float*)d_in[17];
    const float* melW  = (const float*)d_in[18];
    const float* melB  = (const float*)d_in[19];
    const float* wih0 = (const float*)d_in[20];
    const float* whh0 = (const float*)d_in[21];
    const float* bih0 = (const float*)d_in[22];
    const float* bhh0 = (const float*)d_in[23];
    const float* wih1 = (const float*)d_in[24];
    const float* whh1 = (const float*)d_in[25];
    const float* bih1 = (const float*)d_in[26];
    const float* bhh1 = (const float*)d_in[27];
    const float* pw[5], *pb[5], *pg[5], *pbe[5];
    for (int i = 0; i < 5; ++i) {
        pw[i]  = (const float*)d_in[28 + 4 * i];
        pb[i]  = (const float*)d_in[29 + 4 * i];
        pg[i]  = (const float*)d_in[30 + 4 * i];
        pbe[i] = (const float*)d_in[31 + 4 * i];
    }
    float* mel_out = (float*)d_out;                 // first 16*256*80
    float* out2    = mel_out + (size_t)BB * T_MEL * MEL;

    // workspace carve-out
    char* base = (char*)d_ws;
    size_t off = 0;
    auto alloc = [&](size_t bytes) -> char* {
        char* p = base + off;
        off = (off + bytes + 255) & ~(size_t)255;
        return p;
    };
    const size_t PADT = (size_t)BB * PF * HP * WPAD;      // padded activation elems
    bf16_t* Wcat0 = (bf16_t*)alloc((size_t)4096 * 1408 * 2);
    bf16_t* Wcat1 = (bf16_t*)alloc((size_t)4096 * 2048 * 2);
    float*  bias0 = (float*) alloc(4096 * 4);
    float*  bias1 = (float*) alloc(4096 * 4);
    bf16_t* preW1b = (bf16_t*)alloc(256 * 96 * 2);        // K padded 80 -> 96
    bf16_t* preW2b = (bf16_t*)alloc(256 * 256 * 2);
    bf16_t* kvpWb  = (bf16_t*)alloc(128 * 512 * 2);
    bf16_t* kvWb   = (bf16_t*)alloc(256 * 128 * 2);
    bf16_t* pwB[5];
    pwB[0] = (bf16_t*)alloc((size_t)PF * 25 * 2);
    for (int i = 1; i <= 3; ++i) pwB[i] = (bf16_t*)alloc((size_t)PF * PF * 25 * 2);
    pwB[4] = (bf16_t*)alloc((size_t)PF * 25 * 2);
    float* locw    = (float*)alloc(32 * 4);
    float* melApad = (float*)alloc((size_t)4096 * 96 * 4);
    float* pre_out = (float*)alloc((size_t)BB * T_MEL * PRE_OUT * 4);
    float* tmp1    = (float*)alloc((size_t)BB * T_MEL * PRE_OUT * 4);
    float* kvbuf   = (float*)alloc((size_t)BB * T_ENC * 256 * 4);
    float* z0    = (float*)alloc(BB * 1408 * 4);
    float* z1    = (float*)alloc(BB * 2048 * 4);
    float* gates = (float*)alloc(BB * 4096 * 4);
    float* c0s   = (float*)alloc(BB * 1024 * 4);
    float* c1s   = (float*)alloc(BB * 1024 * 4);
    float* pa    = (float*)alloc(BB * T_ENC * 4);
    float* pe    = (float*)alloc(BB * T_ENC * 4);
    bf16_t* actA    = (bf16_t*)alloc(PADT * 2);           // padded
    bf16_t* actB    = (bf16_t*)alloc(PADT * 2);           // padded
    bf16_t* convraw = (bf16_t*)alloc((size_t)BB * PF * HW * 2);
    float*  stats   = (float*)alloc(PF * 2 * 4);

    // ---- weight prep (deterministic, redone each call) ----
    k_cvt_pad<<<256, 256, 0, stream>>>(preW1, preW1b, 256, 80, 96);
    k_cvt<<<256, 256, 0, stream>>>(preW2, preW2b, (size_t)256 * 256);
    k_cvt<<<256, 256, 0, stream>>>(kvpW, kvpWb, (size_t)128 * 512);
    k_cvt<<<256, 256, 0, stream>>>(kvW,  kvWb,  (size_t)256 * 128);
    k_cvt<<<64,  256, 0, stream>>>(pw[0], pwB[0], (size_t)PF * 25);
    for (int i = 1; i <= 3; ++i)
        k_pack_convw<<<8192, 256, 0, stream>>>(pw[i], pwB[i]);
    k_cvt<<<64, 256, 0, stream>>>(pw[4], pwB[4], (size_t)PF * 25);
    k_pack_lstm<<<8192, 256, 0, stream>>>(wih0, whh0, Wcat0, 384, 1024);
    k_pack_lstm<<<8192, 256, 0, stream>>>(wih1, whh1, Wcat1, 1024, 1024);
    k_add2<<<16, 256, 0, stream>>>(bih0, bhh0, bias0, 4096);
    k_add2<<<16, 256, 0, stream>>>(bih1, bhh1, bias1, 4096);
    k_locmean<<<1, 32, 0, stream>>>(locW, locB, locw);
    k_pad_rows<<<1536, 256, 0, stream>>>(mel_seq, melApad, 4096, 80, 96);

    // ---- state init ----
    k_fill<<<(BB * 1408 + 255) / 256, 256, 0, stream>>>(z0, 0.f, BB * 1408);
    k_fill<<<(BB * 2048 + 255) / 256, 256, 0, stream>>>(z1, 0.f, BB * 2048);
    k_fill<<<64, 256, 0, stream>>>(c0s, 0.f, BB * 1024);
    k_fill<<<64, 256, 0, stream>>>(c1s, 0.f, BB * 1024);
    k_fill<<<16, 256, 0, stream>>>(pa, 1.0f / T_ENC, BB * T_ENC);
    k_fill<<<16, 256, 0, stream>>>(pe, 0.f, BB * T_ENC);
    k_fill_bf16<<<65536, 256, 0, stream>>>(actA, PADT);   // zero borders
    k_fill_bf16<<<65536, 256, 0, stream>>>(actB, PADT);

    // ---- precompute: prenet + key/value projections (WMMA GEMMs) ----
    k_gemm_af32<<<dim3(2, 256), 256, 16 * 96 * 2, stream>>>(melApad, preW1b, preB1, tmp1,
                                                            4096, 256, 96, 1);
    k_gemm_af32<<<dim3(2, 256), 256, 16 * 256 * 2, stream>>>(tmp1, preW2b, preB2, pre_out,
                                                             4096, 256, 256, 1);
    k_gemm_af32<<<dim3(1, 256), 256, 16 * 512 * 2, stream>>>(x, kvpWb, kvpB, tmp1,
                                                             4096, 128, 512, 0);
    k_gemm_af32<<<dim3(2, 256), 256, 16 * 128 * 2, stream>>>(tmp1, kvWb, kvB, kvbuf,
                                                             4096, 256, 128, 0);

    // ---- recurrent decode: 256 dependent steps ----
    for (int t = 0; t < T_MEL; ++t) {
        k_attn_step<<<BB, 256, 0, stream>>>(z1, qpW, qpB, qW, qB, kvbuf, locw,
                                            cumW, cumB, pre_out, pa, pe, z0, t);
        k_gemm_af32<<<dim3(32, 1), 256, 16 * 1408 * 2, stream>>>(z0, Wcat0, bias0, gates,
                                                                 16, 4096, 1408, 0);
        k_cell0<<<64, 256, 0, stream>>>(gates, c0s, z1, z0);
        k_gemm_af32<<<dim3(32, 1), 256, 16 * 2048 * 2, stream>>>(z1, Wcat1, bias1, gates,
                                                                 16, 4096, 2048, 0);
        k_cell1_mel<<<BB, 256, 0, stream>>>(gates, c1s, z1, melW, melB, mel_out, t);
    }

    // ---- postnet ----
    k_conv_first<<<65536, 256, 0, stream>>>(mel_out, pwB[0], pb[0], convraw);
    k_bnstats<<<PF, 256, 0, stream>>>(convraw, stats, PF);
    k_bnact_pad<<<65536, 256, 0, stream>>>(convraw, actA, stats, pg[0], pbe[0]);

    bf16_t* cin = actA;
    bf16_t* cout = actB;
    const int PATCH_LDS = PF * 5 * 20 * 2;                // 102400 bytes
    for (int layer = 1; layer <= 3; ++layer) {
        k_conv5x5_wmma<<<dim3(4, CH * 16, BB), 256, PATCH_LDS, stream>>>(cin, pwB[layer],
                                                                         pb[layer], convraw);
        k_bnstats<<<PF, 256, 0, stream>>>(convraw, stats, PF);
        k_bnact_pad<<<65536, 256, 0, stream>>>(convraw, cout, stats, pg[layer], pbe[layer]);
        bf16_t* tmp = cin; cin = cout; cout = tmp;
    }
    k_conv_last<<<1280, 256, 0, stream>>>(cin, pwB[4], pb[4], convraw);  // C=1 layout
    k_bnstats<<<1, 256, 0, stream>>>(convraw, stats, 1);
    k_final_add<<<1280, 256, 0, stream>>>(convraw, stats, pg[4], pbe[4], mel_out, out2);

    (void)in_sizes; (void)n_in; (void)out_size; (void)ws_size;
}
